// MessagePassing_28389733826999
// MI455X (gfx1250) — matrix operand order repaired and checked by reading, not compiled, not run on hardware
//
#include <hip/hip_runtime.h>

// ---------------------------------------------------------------------------
// Problem constants (match reference)
// ---------------------------------------------------------------------------
static constexpr int NN  = 10000;   // num nodes
static constexpr int NE  = 160000;  // num edges
static constexpr int ND  = 256;     // node dim
static constexpr int ER  = 64;      // edge radial dim
static constexpr int EA  = 32;      // edge angular dim
static constexpr int EC  = ER + EA; // 96
static constexpr int AC  = EC + ND; // 352

typedef _Float16 v16h __attribute__((ext_vector_type(16)));
typedef _Float16 v8h  __attribute__((ext_vector_type(8)));
typedef _Float16 v4h  __attribute__((ext_vector_type(4)));
typedef float    v8f  __attribute__((ext_vector_type(8)));

__device__ __forceinline__ float sigmoidf_(float x) {
    return 1.0f / (1.0f + __expf(-x));
}

// ---------------------------------------------------------------------------
// Elementwise helper kernels
// ---------------------------------------------------------------------------
// Convert f32 weight [Ktrue x Ntrue] into zero-padded f16 [Kp x Np]
__global__ void convert_pad_kernel(const float* __restrict__ in,
                                   _Float16* __restrict__ out,
                                   int Ktrue, int Ntrue, int Np, int total) {
    int i = blockIdx.x * 256 + threadIdx.x;
    if (i < total) {
        int k = i / Np, n = i % Np;
        _Float16 v = (_Float16)0.0f;
        if (n < Ntrue && k < Ktrue) v = (_Float16)in[k * Ntrue + n];
        out[i] = v;
    }
}

__global__ void fill_zero_kernel(float* __restrict__ p, int n) {
    int i = blockIdx.x * 256 + threadIdx.x;
    if (i < n) p[i] = 0.0f;
}

// nfu = sigmoid(t1 + t2) + node_features   (biases already folded into t1/t2)
__global__ void nfu_kernel(const float* __restrict__ t1,
                           const float* __restrict__ t2,
                           const float* __restrict__ node,
                           float* __restrict__ out, int n) {
    int i = blockIdx.x * 256 + threadIdx.x;
    if (i < n) out[i] = sigmoidf_(t1[i] + t2[i]) + node[i];
}

// ---------------------------------------------------------------------------
// Generic WMMA GEMM:  out = EPI( A(rows x Kp) @ B(Kp x Np) + bias )
// Block = 128 threads (4 waves). Block tile = 32 rows x 64 cols; each wave
// owns 2 stacked 16x16 C tiles (v_wmma_f32_16x16x32_f16), K staged through
// double-buffered LDS as f16, one barrier per 32-wide K chunk.
//
//  AMODE 0: dense f32 A (row stride ldA, zero-padded to Kp)
//  AMODE 1: gathered concat(nf[src[src[row]]], er[row], ea[row])
//  AMODE 2: gathered concat(nf[gidx[row]], er, ea)  (+gidx2 tile when DUAL)
//  AMODE 4: dense f16 A (row stride ldA, zero-padded to Kp)
//
//  EPI 0: store f32           EPI 1: relu -> f32
//  EPI 3: sigmoid + atomicAdd into aggOut[dst[row]*Np + col]
//  EPI 4: relu -> f16
//  EPI 5 (DUAL): sigmoid(acc0+b0) + sigmoid(acc1+b1) -> f16
// ---------------------------------------------------------------------------
template <int AMODE, int EPI, bool DUAL>
__global__ __launch_bounds__(128) void gemm_wmma_kernel(
    const float* __restrict__ Af32, const _Float16* __restrict__ Af16,
    const float* __restrict__ nf, const float* __restrict__ er,
    const float* __restrict__ ea, const int* __restrict__ gidx,
    const int* __restrict__ gidx2, const _Float16* __restrict__ B0,
    const float* __restrict__ bias0, const _Float16* __restrict__ B1,
    const float* __restrict__ bias1, float* __restrict__ outF32,
    _Float16* __restrict__ outF16, float* __restrict__ aggOut,
    const int* __restrict__ dstIdx, int rows, int Kp, int ldA, int Nout,
    int Np) {

    constexpr int NT = DUAL ? 2 : 1;
    __shared__ _Float16 Alds[2][NT][32][48];  // [buf][tile][row][K-chunk(+pad)]
    __shared__ int sIdx[2][32];
    __shared__ int sDst[32];

    const int tid  = threadIdx.x;
    const int lane = tid & 31;
    const int wave = tid >> 5;
    const int m = lane & 15;
    const int g = lane >> 4;
    const int rowTile = blockIdx.x;
    const int rowBase = rowTile * 32;
    const int n0 = blockIdx.y * 64 + wave * 16;
    const bool colOK = (n0 < Np);
    const int n0c = colOK ? n0 : 0;  // clamped for safe (discarded) loads

    // ---- per-block gather-index / scatter-index pre-pass -----------------
    if (tid < 32) {
        int grow = rowBase + tid;
        int gc = (grow < rows) ? grow : (rows - 1);
        if constexpr (AMODE == 1) {
            int s1 = gidx[gc];
            sIdx[0][tid] = gidx[s1];  // src[src[e]] quirk
        } else if constexpr (AMODE == 2) {
            sIdx[0][tid] = gidx[gc];
            if constexpr (DUAL) sIdx[1][tid] = gidx2[gc];
        }
        if constexpr (EPI == 3) sDst[tid] = dstIdx[gc];
    }
    __syncthreads();

    // ---- cooperative fill of one 32x32 f16 chunk (per tile) --------------
    // thread t handles rows (t>>3) and (t>>3)+16, 4 contiguous K each.
    auto fillChunk = [&](int buf, int k0) {
        const int r0 = tid >> 3;
        const int kk = (tid & 7) * 4;
#pragma unroll
        for (int s = 0; s < 2; ++s) {
            const int r = r0 + 16 * s;
            const int grow = rowBase + r;
            const int rowc = (grow < rows) ? grow : (rows - 1);
#pragma unroll
            for (int t = 0; t < NT; ++t) {
                if constexpr (AMODE == 4) {
                    v4h v = *(const v4h*)(Af16 + (size_t)rowc * ldA + k0 + kk);
                    *(v4h*)&Alds[buf][t][r][kk] = v;
                } else {
                    float4 v;
                    if constexpr (AMODE == 0) {
                        v = *(const float4*)(Af32 + (size_t)rowc * ldA + k0 + kk);
                    } else {
                        // chunk-uniform region select (k0 multiple of 32;
                        // boundaries 256/320 are multiples of 32)
                        if (k0 < ND) {
                            v = *(const float4*)(nf + (size_t)sIdx[t][r] * ND +
                                                 k0 + kk);
                        } else if (k0 < ND + ER) {
                            v = *(const float4*)(er + (size_t)rowc * ER +
                                                 (k0 - ND) + kk);
                        } else {
                            v = *(const float4*)(ea + (size_t)rowc * EA +
                                                 (k0 - ND - ER) + kk);
                        }
                    }
                    v4h h;
                    h[0] = (_Float16)v.x;
                    h[1] = (_Float16)v.y;
                    h[2] = (_Float16)v.z;
                    h[3] = (_Float16)v.w;
                    *(v4h*)&Alds[buf][t][r][kk] = h;
                }
            }
        }
    };

    // A fragment: ISA 16-bit 16x32 A layout -> two contiguous 16B LDS reads
    auto loadAfrag = [&](int buf, int t, int sub) -> v16h {
        const _Float16* p = &Alds[buf][t][sub * 16 + m][g * 8];
        v8h lo = *(const v8h*)p;         // K = 8g .. 8g+7
        v8h hi = *(const v8h*)(p + 16);  // K = 16+8g .. 16+8g+7
        v16h a;
#pragma unroll
        for (int i = 0; i < 8; ++i) { a[i] = lo[i]; a[8 + i] = hi[i]; }
        return a;
    };

    // B fragment: lane holds column N = n0c+m, K = 8g.. and 16+8g.. (matches A)
    auto loadBfrag = [&](const _Float16* B, int k0) -> v16h {
        const _Float16* p = B + (size_t)(k0 + g * 8) * Np + n0c + m;
        v16h b;
#pragma unroll
        for (int i = 0; i < 8; ++i) {
            b[i] = p[(size_t)i * Np];
            b[8 + i] = p[(size_t)(16 + i) * Np];
        }
        return b;
    };

    v8f c00 = {}, c01 = {};  // tile 0, row halves 0/1
    v8f c10 = {}, c11 = {};  // tile 1 (DUAL)

    fillChunk(0, 0);
    __syncthreads();

    const int nch = Kp >> 5;
    for (int c = 0; c < nch; ++c) {
        const int buf = c & 1;
        const int k0 = c << 5;

        v16h b0 = loadBfrag(B0, k0);
        v16h a00 = loadAfrag(buf, 0, 0);
        v16h a01 = loadAfrag(buf, 0, 1);
        v16h b1, a10, a11;
        if constexpr (DUAL) {
            b1  = loadBfrag(B1, k0);
            a10 = loadAfrag(buf, 1, 0);
            a11 = loadAfrag(buf, 1, 1);
        }

        if (c + 1 < nch) fillChunk(buf ^ 1, (c + 1) << 5);

        c00 = __builtin_amdgcn_wmma_f32_16x16x32_f16(false, a00, false, b0,
                                                     (short)0, c00, false, false);
        c01 = __builtin_amdgcn_wmma_f32_16x16x32_f16(false, a01, false, b0,
                                                     (short)0, c01, false, false);
        if constexpr (DUAL) {
            c10 = __builtin_amdgcn_wmma_f32_16x16x32_f16(
                false, a10, false, b1, (short)0, c10, false, false);
            c11 = __builtin_amdgcn_wmma_f32_16x16x32_f16(
                false, a11, false, b1, (short)0, c11, false, false);
        }
        __syncthreads();
    }

    // ---- epilogue --------------------------------------------------------
    if (colOK) {
        const int col = n0 + m;
        const float bv0 = (col < Nout) ? bias0[col] : 0.0f;
        float bv1 = 0.0f;
        if constexpr (DUAL) bv1 = (col < Nout) ? bias1[col] : 0.0f;

#pragma unroll
        for (int sub = 0; sub < 2; ++sub) {
            v8f acc0 = sub ? c01 : c00;
            v8f acc1 = sub ? c11 : c10;
#pragma unroll
            for (int vi = 0; vi < 8; ++vi) {
                int rloc = sub * 16 + vi + 8 * g;  // row within 32-row tile
                int row = rowBase + rloc;
                if (row < rows) {
                    float x = acc0[vi] + bv0;
                    if constexpr (EPI == 0) {
                        outF32[(size_t)row * Np + col] = x;
                    } else if constexpr (EPI == 1) {
                        outF32[(size_t)row * Np + col] = fmaxf(x, 0.0f);
                    } else if constexpr (EPI == 3) {
                        atomicAdd(&aggOut[(size_t)sDst[rloc] * Np + col],
                                  sigmoidf_(x));
                    } else if constexpr (EPI == 4) {
                        outF16[(size_t)row * Np + col] =
                            (_Float16)fmaxf(x, 0.0f);
                    } else if constexpr (EPI == 5) {
                        float x1 = acc1[vi] + bv1;
                        outF16[(size_t)row * Np + col] =
                            (_Float16)(sigmoidf_(x) + sigmoidf_(x1));
                    }
                }
            }
        }
    }
}

// ---------------------------------------------------------------------------
// Host orchestration
// ---------------------------------------------------------------------------
static inline int cdiv_(int a, int b) { return (a + b - 1) / b; }
static inline int pad32_(int x) { return (x + 31) & ~31; }

extern "C" void kernel_launch(void* const* d_in, const int* in_sizes, int n_in,
                              void* d_out, int out_size, void* d_ws,
                              size_t ws_size, hipStream_t stream) {
    (void)in_sizes; (void)n_in; (void)out_size; (void)ws_size;

    const float* node = (const float*)d_in[0];
    const float* erad = (const float*)d_in[1];
    const float* eang = (const float*)d_in[2];
    const int*   eidx = (const int*)d_in[3];
    const int*   src  = eidx;
    const int*   dst  = eidx + NE;

    const float* mlp1_w = (const float*)d_in[4];
    const float* mlp1_b = (const float*)d_in[5];
    const float* m2w[4] = {(const float*)d_in[6],  (const float*)d_in[8],
                           (const float*)d_in[10], (const float*)d_in[12]};
    const float* m2b[4] = {(const float*)d_in[7],  (const float*)d_in[9],
                           (const float*)d_in[11], (const float*)d_in[13]};
    const float* nls_w = (const float*)d_in[14];
    const float* nls_b = (const float*)d_in[15];
    const float* nln_w = (const float*)d_in[16];
    const float* nln_b = (const float*)d_in[17];
    const float* em1_w = (const float*)d_in[18];
    const float* em1_b = (const float*)d_in[19];
    const float* e2w[4] = {(const float*)d_in[20], (const float*)d_in[22],
                           (const float*)d_in[24], (const float*)d_in[26]};
    const float* e2b[4] = {(const float*)d_in[21], (const float*)d_in[23],
                           (const float*)d_in[25], (const float*)d_in[27]};

    float* nfu = (float*)d_out;                    // [N, ND]
    float* efu = (float*)d_out + (size_t)NN * ND;  // [E, EC]

    // ---- workspace layout ------------------------------------------------
    char* ws = (char*)d_ws;
    size_t off = 0;
    auto take = [&](size_t bytes) -> char* {
        char* p = ws + off;
        off += (bytes + 255) & ~(size_t)255;
        return p;
    };

    // True dims of MLP chains
    const int m2d[5] = {AC, 208, 64, ND / 2, ND};
    const int e2d[5] = {AC, 188, 24, EC / 2, EC};

    // Padded f16 weights (Kp x Np, zero padded)
    auto wtake = [&](int Kt, int Nt) -> _Float16* {
        return (_Float16*)take((size_t)pad32_(Kt) * pad32_(Nt) * 2);
    };
    _Float16* w16_mlp1 = wtake(AC, AC);
    _Float16* w16_m2[4];
    for (int i = 0; i < 4; ++i) w16_m2[i] = wtake(m2d[i], m2d[i + 1]);
    _Float16* w16_nls = wtake(ND, ND);
    _Float16* w16_nln = wtake(ND, ND);
    _Float16* w16_em1 = wtake(AC, AC);
    _Float16* w16_e2[4];
    for (int i = 0; i < 4; ++i) w16_e2[i] = wtake(e2d[i], e2d[i + 1]);

    size_t scratch_base = off;
    // node phase buffers (f32, padded strides)
    float* agg  = (float*)take((size_t)NN * AC * 4);          // stride 352
    float* h0   = (float*)take((size_t)NN * pad32_(208) * 4); // stride 224
    float* h1   = (float*)take((size_t)NN * 64 * 4);
    float* h2n  = (float*)take((size_t)NN * 128 * 4);
    float* mbuf = (float*)take((size_t)NN * ND * 4);
    float* t1   = (float*)take((size_t)NN * ND * 4);
    float* t2   = (float*)take((size_t)NN * ND * 4);
    // edge phase buffers (f16) — alias node phase (stream-ordered)
    off = scratch_base;
    _Float16* bufH2 = (_Float16*)take((size_t)NE * AC * 2);          // 352
    _Float16* eb0   = (_Float16*)take((size_t)NE * pad32_(188) * 2); // 192
    _Float16* eb1   = (_Float16*)take((size_t)NE * pad32_(24) * 2);  // 32
    _Float16* eb2   = (_Float16*)take((size_t)NE * pad32_(48) * 2);  // 64

    // ---- weight conversion f32 -> padded f16 -----------------------------
    auto conv = [&](const float* in, _Float16* out, int Kt, int Nt) {
        int total = pad32_(Kt) * pad32_(Nt);
        convert_pad_kernel<<<cdiv_(total, 256), 256, 0, stream>>>(
            in, out, Kt, Nt, pad32_(Nt), total);
    };
    conv(mlp1_w, w16_mlp1, AC, AC);
    for (int i = 0; i < 4; ++i) conv(m2w[i], w16_m2[i], m2d[i], m2d[i + 1]);
    conv(nls_w, w16_nls, ND, ND);
    conv(nln_w, w16_nln, ND, ND);
    conv(em1_w, w16_em1, AC, AC);
    for (int i = 0; i < 4; ++i) conv(e2w[i], w16_e2[i], e2d[i], e2d[i + 1]);

    // ---- 1) msg = sigmoid(concat(nf[src[src]],er,ea) @ mlp1 + b), scatter
    fill_zero_kernel<<<cdiv_(NN * AC, 256), 256, 0, stream>>>(agg, NN * AC);
    gemm_wmma_kernel<1, 3, false>
        <<<dim3(cdiv_(NE, 32), cdiv_(AC, 64)), 128, 0, stream>>>(
            nullptr, nullptr, node, erad, eang, src, nullptr, w16_mlp1, mlp1_b,
            nullptr, nullptr, nullptr, nullptr, agg, dst, NE, AC, 0, AC, AC);

    // ---- 2) node MLP chain: 352 -> 208 -> 64 -> 128 -> 256 ---------------
    auto gemm_node = [&](const float* A, const _Float16* B, const float* b,
                         float* out, int K, int Nout, bool relu) {
        int Kp = pad32_(K), Np = pad32_(Nout);
        dim3 grid(cdiv_(NN, 32), cdiv_(Np, 64));
        if (relu)
            gemm_wmma_kernel<0, 1, false><<<grid, 128, 0, stream>>>(
                A, nullptr, nullptr, nullptr, nullptr, nullptr, nullptr, B, b,
                nullptr, nullptr, out, nullptr, nullptr, nullptr, NN, Kp, Kp,
                Nout, Np);
        else
            gemm_wmma_kernel<0, 0, false><<<grid, 128, 0, stream>>>(
                A, nullptr, nullptr, nullptr, nullptr, nullptr, nullptr, B, b,
                nullptr, nullptr, out, nullptr, nullptr, nullptr, NN, Kp, Kp,
                Nout, Np);
    };
    gemm_node(agg, w16_m2[0], m2b[0], h0,   AC,  208, true);
    gemm_node(h0,  w16_m2[1], m2b[1], h1,   208, 64,  true);
    gemm_node(h1,  w16_m2[2], m2b[2], h2n,  64,  128, true);
    gemm_node(h2n, w16_m2[3], m2b[3], mbuf, 128, ND,  false);

    // ---- 3) nfu = sigmoid(node@nls+b + m@nln+b) + node -------------------
    gemm_node(node, w16_nls, nls_b, t1, ND, ND, false);
    gemm_node(mbuf, w16_nln, nln_b, t2, ND, ND, false);
    nfu_kernel<<<cdiv_(NN * ND, 256), 256, 0, stream>>>(t1, t2, node, nfu,
                                                        NN * ND);

    // ---- 4) h2 = sigmoid(s_in@em1+b) + sigmoid(t_in@mlp1+b)  (dual) ------
    gemm_wmma_kernel<2, 5, true>
        <<<dim3(cdiv_(NE, 32), cdiv_(AC, 64)), 128, 0, stream>>>(
            nullptr, nullptr, nfu, erad, eang, src, dst, w16_em1, em1_b,
            w16_mlp1, mlp1_b, nullptr, bufH2, nullptr, nullptr, NE, AC, 0, AC,
            AC);

    // ---- 5) edge MLP chain: 352 -> 188 -> 24 -> 48 -> 96 -----------------
    auto gemm_edge = [&](const _Float16* A, const _Float16* B, const float* b,
                         _Float16* out16, float* out32, int K, int Nout,
                         bool relu16) {
        int Kp = pad32_(K), Np = pad32_(Nout);
        dim3 grid(cdiv_(NE, 32), cdiv_(Np, 64));
        if (relu16)
            gemm_wmma_kernel<4, 4, false><<<grid, 128, 0, stream>>>(
                nullptr, A, nullptr, nullptr, nullptr, nullptr, nullptr, B, b,
                nullptr, nullptr, nullptr, out16, nullptr, nullptr, NE, Kp, Kp,
                Nout, Np);
        else
            gemm_wmma_kernel<4, 0, false><<<grid, 128, 0, stream>>>(
                nullptr, A, nullptr, nullptr, nullptr, nullptr, nullptr, B, b,
                nullptr, nullptr, out32, nullptr, nullptr, nullptr, NE, Kp, Kp,
                Nout, Np);
    };
    gemm_edge(bufH2, w16_e2[0], e2b[0], eb0, nullptr, AC,  188, true);
    gemm_edge(eb0,   w16_e2[1], e2b[1], eb1, nullptr, 188, 24,  true);
    gemm_edge(eb1,   w16_e2[2], e2b[2], eb2, nullptr, 24,  48,  true);
    gemm_edge(eb2,   w16_e2[3], e2b[3], nullptr, efu, 48,  EC,  false);
}
